// LSTM_50491635532240
// MI455X (gfx1250) — matrix operand order repaired and checked by reading, not compiled, not run on hardware
//
#include <hip/hip_runtime.h>
#include <hip/hip_bf16.h>
#include <cstdint>

// Problem dims (fixed by the reference)
#define BATCH 64
#define SEQ   1024
#define DIN_  256
#define HID   512
#define DOUT_ 256

typedef __attribute__((ext_vector_type(16))) __bf16 v16bf;
typedef __attribute__((ext_vector_type(8)))  float  v8f;

// Opportunistic async global->LDS path (ASYNCcnt); falls back to
// load+ds_store staging if the builtins are absent on this toolchain.
#if defined(__has_builtin)
#  if __has_builtin(__builtin_amdgcn_global_load_async_to_lds_b128) && \
      __has_builtin(__builtin_amdgcn_s_wait_asynccnt)
#    define HAVE_ASYNC_LDS 1
#  endif
#endif
#ifndef HAVE_ASYNC_LDS
#  define HAVE_ASYNC_LDS 0
#endif

#if HAVE_ASYNC_LDS
// Probe-derived signature: arg0 is 'v4i addrspace(1)*' (clang prints AS1 as
// __device__ in HIP diagnostics); arg1 is the LDS-side pointer.
typedef int v4i_ __attribute__((ext_vector_type(4)));
typedef __attribute__((address_space(1))) v4i_* as1_v4i;
typedef __attribute__((address_space(3))) v4i_* as3_v4i;
#endif

__device__ __forceinline__ float sigmoidf_(float x) {
    return 1.0f / (1.0f + __expf(-x));
}

// ---------------------------------------------------------------------------
// Device-wide barrier (persistent-kernel sync). All blocks co-resident
// (<=16 blocks on a many-WGP part). counter+generation protocol.
// ---------------------------------------------------------------------------
__device__ __forceinline__ void grid_sync(unsigned* cnt, unsigned* gen, unsigned nblocks) {
    __threadfence();          // release my stores to device scope
    __syncthreads();
    if (threadIdx.x == 0) {
        unsigned g = __hip_atomic_load(gen, __ATOMIC_ACQUIRE, __HIP_MEMORY_SCOPE_AGENT);
        unsigned a = __hip_atomic_fetch_add(cnt, 1u, __ATOMIC_ACQ_REL, __HIP_MEMORY_SCOPE_AGENT);
        if (a == nblocks - 1u) {
            __hip_atomic_store(cnt, 0u, __ATOMIC_RELAXED, __HIP_MEMORY_SCOPE_AGENT);
            __hip_atomic_store(gen, g + 1u, __ATOMIC_RELEASE, __HIP_MEMORY_SCOPE_AGENT);
        } else {
            while (__hip_atomic_load(gen, __ATOMIC_ACQUIRE, __HIP_MEMORY_SCOPE_AGENT) == g) {
                __builtin_amdgcn_s_sleep(8);
            }
        }
    }
    __syncthreads();
    __threadfence();          // acquire: invalidate near caches before reading peers' h
}

// ---------------------------------------------------------------------------
// Weight packer: fp32 row-major W[K][N] -> bf16 fragment-ready layout.
// B-fragment layout (ISA 7.12.2, 16-bit B 32x16): lane%16 <-> N within a
// 16-N tile, 16 packed halves per lane <-> K (table order) within a 32-K block.
// packed index p = ((ctile*KT + ktile)*32 + lane)*16 + slot
//   K = ktile*32 + 8*(lane/16) + slot%8 + 16*(slot/8) ; N = ctile*16 + lane%16
// ---------------------------------------------------------------------------
__global__ void __launch_bounds__(256) pack_w_kernel(const float* __restrict__ W,
                                                     __bf16* __restrict__ dst,
                                                     int K, int N) {
    int total = K * N;
    int KT = K >> 5;
    for (int p = blockIdx.x * blockDim.x + threadIdx.x; p < total;
         p += gridDim.x * blockDim.x) {
        int slot = p & 15;
        int rest = p >> 4;
        int lane = rest & 31;
        int tile = rest >> 5;
        int kt = tile % KT;
        int ct = tile / KT;
        int k = kt * 32 + ((lane >> 4) << 3) + (slot & 7) + ((slot >> 3) << 4);
        int n = ct * 16 + (lane & 15);
        dst[p] = (__bf16)W[(size_t)k * N + n];
    }
}

__global__ void __launch_bounds__(256) combine_bias_kernel(const float* __restrict__ a,
                                                           const float* __restrict__ b,
                                                           float* __restrict__ dst, int n) {
    int i = blockIdx.x * blockDim.x + threadIdx.x;
    if (i < n) dst[i] = a[i] + b[i];
}

__global__ void __launch_bounds__(256) init_state_kernel(float* __restrict__ c_buf,
                                                         __bf16* __restrict__ h_buf,
                                                         int nc, int nh,
                                                         unsigned* __restrict__ sync) {
    int i = blockIdx.x * blockDim.x + threadIdx.x;
    if (i < 2) sync[i] = 0u;
    for (int j = i; j < nc; j += gridDim.x * blockDim.x) c_buf[j] = 0.0f;
    for (int j = i; j < nh; j += gridDim.x * blockDim.x) h_buf[j] = (__bf16)0.0f;
}

// ---------------------------------------------------------------------------
// Hoisted input projection: pre-activations = A @ W_ih + (b_ih + b_hh).
// One wave per 16x16 output tile; K-loop of v_wmma_f32_16x16x32_bf16.
// Output is stored in WMMA C-fragment-native layout so the recurrence can
// reload its accumulators with one contiguous 32B v8f load per lane:
//   xgf[((t*npairs + pair)*4 + gate)*256 + lane*8 + r]
// where pair = m*CT + c  (m: batch tile, c: per-gate column tile).
// layer0: A = x (fp32, (B,T,DIN) layout, converted to bf16 on the fly)
// layer1: A = h1seq (bf16, [t][b][HID])
// ---------------------------------------------------------------------------
__global__ void __launch_bounds__(256) input_gemm_kernel(const float* __restrict__ x_f32,
                                                         const __bf16* __restrict__ a_bf16,
                                                         const __bf16* __restrict__ wpack,
                                                         const float* __restrict__ bias,
                                                         float* __restrict__ xgf,
                                                         int Ksz, int Nsz, int npairs,
                                                         int is_layer0) {
    const int lane = threadIdx.x & 31;
    const int wave = blockIdx.x * 8 + (threadIdx.x >> 5);
    const int Ntiles = Nsz >> 4;
    const int CT = Ntiles >> 2;                 // per-gate column tiles
    const int mtile = wave / Ntiles;
    const int ctile = wave % Ntiles;
    const int KT = Ksz >> 5;

    const int ncol  = (ctile << 4) + (lane & 15);
    const int half8 = (lane >> 4) << 3;           // A-layout: K sub-offset per half-wave
    const int arow  = (mtile << 4) + (lane & 15); // A-layout: M = lane%16

    v8f acc;
    const float bv = bias[ncol];
#pragma unroll
    for (int r = 0; r < 8; ++r) acc[r] = bv;

    const float*  xrow = nullptr;
    const __bf16* hrow = nullptr;
    if (is_layer0) {
        // output row r = t*B + b reads x[b][t][:]
        xrow = x_f32 + (size_t)(arow % BATCH) * SEQ * DIN_ + (size_t)(arow / BATCH) * DIN_;
    } else {
        hrow = a_bf16 + (size_t)arow * Ksz;
    }

    for (int kb = 0; kb < KT; ++kb) {
        v16bf a;
        const int k0 = kb * 32 + half8;
        if (is_layer0) {
#pragma unroll
            for (int g = 0; g < 2; ++g)
#pragma unroll
                for (int i = 0; i < 8; ++i)
                    a[g * 8 + i] = (__bf16)xrow[k0 + g * 16 + i];
        } else {
#pragma unroll
            for (int g = 0; g < 2; ++g)
#pragma unroll
                for (int i = 0; i < 8; ++i)
                    a[g * 8 + i] = hrow[k0 + g * 16 + i];
        }
        const v16bf b =
            *(const v16bf*)(wpack + (((size_t)ctile * KT + kb) * 32 + lane) * 16);
        acc = __builtin_amdgcn_wmma_f32_16x16x32_bf16(false, a, false, b,
                                                      (short)0, acc, false, false);
    }

    // fragment-native store: t = mtile/4, m = mtile%4, gate = ctile/CT, c = ctile%CT
    const int t    = mtile >> 2;
    const int m    = mtile & 3;
    const int gate = ctile / CT;
    const int c    = ctile % CT;
    float* dst = xgf + (((size_t)t * npairs + (m * CT + c)) * 4 + gate) * 256 + (lane << 3);
    *(v8f*)dst = acc;
}

// ---------------------------------------------------------------------------
// Persistent recurrence kernel. Each wave owns one (batch-tile m, h-column
// tile c) pair and computes all 4 gate tiles (i,f,g,o) for it so the cell
// update is wave-local. All 8 waves of a block share the same batch tile, so
// the h_t tile (16 x Hdim bf16) is staged into LDS once per step (async
// global->LDS when available) and WMMA A-fragments are fed from LDS.
// h is double-buffered (parity t); c stays fp32 and is wave-exclusive.
// Next step's pre-activation fragments are prefetched (global_prefetch) to
// overlap their HBM latency with the current step's WMMA work.
// One device barrier per step.
// ---------------------------------------------------------------------------
__global__ void __launch_bounds__(256) lstm_rec_kernel(const float* __restrict__ xgf,
                                                       const __bf16* __restrict__ wpack,
                                                       __bf16* __restrict__ h_buf,   // 2 * B * Hdim
                                                       float* __restrict__ c_buf,    // B * Hdim
                                                       __bf16* __restrict__ h1seq,   // [t][b][Hdim] (layer 0)
                                                       float* __restrict__ fout,     // (B,T,Hdim)  (layer 1)
                                                       int Hdim, int npairs,
                                                       unsigned* __restrict__ sync,
                                                       int nblocks, int is_final) {
    __shared__ __align__(16) __bf16 smem[16 * HID];   // one 16-row h tile (<=16KB)

    const int lane = threadIdx.x & 31;
    const int wave = blockIdx.x * 8 + (threadIdx.x >> 5);
    const int CT = Hdim >> 4;   // 16-col tiles per gate
    const int KT = Hdim >> 5;   // 32-K blocks
    const bool active = wave < npairs;
    const int c = active ? (wave % CT) : 0;
    const int mb = (blockIdx.x * 8) / CT;      // batch tile shared by the whole block
    const int rbase = (lane >> 4) << 3;
    const int ncol  = (c << 4) + (lane & 15);
    const int half8 = (lane >> 4) << 3;
    const int BH = BATCH * Hdim;
    const int nchunks = (16 * Hdim) >> 3;      // 16B chunks in the h tile

    for (int t = 0; t < SEQ; ++t) {
        const __bf16* hsrc = h_buf + (size_t)(t & 1) * BH;         // h_t
        __bf16*       hdst = h_buf + (size_t)((t + 1) & 1) * BH;   // h_{t+1}

        // ---- stage h_t tile (16 rows x Hdim) into LDS ----
        {
            const __bf16* gsrc = hsrc + (size_t)mb * 16 * Hdim;
            for (int i = threadIdx.x; i < nchunks; i += 256) {
#if HAVE_ASYNC_LDS
                __builtin_amdgcn_global_load_async_to_lds_b128(
                    (as1_v4i)(uintptr_t)(gsrc + (size_t)i * 8),
                    (as3_v4i)(void*)(smem + i * 8), 0, 0);
#else
                ((uint4*)smem)[i] = ((const uint4*)gsrc)[i];
#endif
            }
#if HAVE_ASYNC_LDS
            __builtin_amdgcn_s_wait_asynccnt(0);
#endif
        }
        __syncthreads();

        if (active) {
            // C-init: hoisted pre-activations in fragment-native layout
            const float* xgt =
                xgf + (((size_t)t * npairs + wave) * 4) * 256 + (lane << 3);
            v8f acc[4];
#pragma unroll
            for (int g = 0; g < 4; ++g)
                acc[g] = *(const v8f*)(xgt + g * 256);

            // prefetch next step's fragments (global_prefetch_b8) so the HBM
            // latency overlaps with this step's WMMA work
            {
                const int tn = (t + 1 < SEQ) ? (t + 1) : t;
                const float* nxt =
                    xgf + (((size_t)tn * npairs + wave) * 4) * 256 + (lane << 3);
#pragma unroll
                for (int g = 0; g < 4; ++g)
                    __builtin_prefetch(nxt + g * 256, 0, 1);
            }

            // gates += h_t @ W_hh   (A fragment from LDS, shared by 4 gates)
            const __bf16* hrow = smem + (lane & 15) * Hdim + half8;
            for (int kb = 0; kb < KT; ++kb) {
                v16bf a;
                const __bf16* p = hrow + kb * 32;
#pragma unroll
                for (int gg = 0; gg < 2; ++gg)
#pragma unroll
                    for (int i = 0; i < 8; ++i)
                        a[gg * 8 + i] = p[gg * 16 + i];
#pragma unroll
                for (int g = 0; g < 4; ++g) {
                    const v16bf b = *(const v16bf*)(
                        wpack + (((size_t)(g * CT + c) * KT + kb) * 32 + lane) * 16);
                    acc[g] = __builtin_amdgcn_wmma_f32_16x16x32_bf16(
                        false, a, false, b, (short)0, acc[g], false, false);
                }
            }

            // elementwise cell update (torch chunk order: i, f, g, o)
#pragma unroll
            for (int r = 0; r < 8; ++r) {
                const int brow = mb * 16 + rbase + r;
                const float iv = sigmoidf_(acc[0][r]);
                const float fv = sigmoidf_(acc[1][r]);
                const float gv = tanhf(acc[2][r]);
                const float ov = sigmoidf_(acc[3][r]);
                const size_t idx = (size_t)brow * Hdim + ncol;
                const float cn = fv * c_buf[idx] + iv * gv;
                const float hn = ov * tanhf(cn);
                c_buf[idx] = cn;
                hdst[idx]  = (__bf16)hn;
                if (is_final)
                    fout[((size_t)brow * SEQ + t) * Hdim + ncol] = hn;
                else
                    h1seq[((size_t)t * BATCH + brow) * Hdim + ncol] = (__bf16)hn;
            }
        }
        grid_sync(sync, sync + 1, (unsigned)nblocks);
    }
}

// ---------------------------------------------------------------------------
// Host launcher
// ---------------------------------------------------------------------------
extern "C" void kernel_launch(void* const* d_in, const int* in_sizes, int n_in,
                              void* d_out, int out_size, void* d_ws, size_t ws_size,
                              hipStream_t stream) {
    (void)in_sizes; (void)n_in; (void)out_size; (void)ws_size;

    const float* x     = (const float*)d_in[0];
    const float* W_ih0 = (const float*)d_in[1];
    const float* b_ih0 = (const float*)d_in[2];
    const float* W_hh0 = (const float*)d_in[3];
    const float* b_hh0 = (const float*)d_in[4];
    const float* W_ih1 = (const float*)d_in[5];
    const float* b_ih1 = (const float*)d_in[6];
    const float* W_hh1 = (const float*)d_in[7];
    const float* b_hh1 = (const float*)d_in[8];
    float* out = (float*)d_out;

    char* ws = (char*)d_ws;
    size_t off = 0;
    auto alloc = [&](size_t bytes) -> char* {
        char* p = ws + off;
        off = (off + bytes + 255) & ~(size_t)255;
        return p;
    };

    __bf16* wp_ih0 = (__bf16*)alloc((size_t)DIN_ * 4 * HID   * 2);  // 1 MB
    __bf16* wp_hh0 = (__bf16*)alloc((size_t)HID  * 4 * HID   * 2);  // 2 MB
    __bf16* wp_ih1 = (__bf16*)alloc((size_t)HID  * 4 * DOUT_ * 2);  // 1 MB
    __bf16* wp_hh1 = (__bf16*)alloc((size_t)DOUT_* 4 * DOUT_ * 2);  // 0.5 MB
    float*  bias0  = (float*) alloc(4 * HID   * 4);
    float*  bias1  = (float*) alloc(4 * DOUT_ * 4);
    unsigned* sync = (unsigned*)alloc(256);
    __bf16* h_buf  = (__bf16*)alloc((size_t)2 * BATCH * HID * 2);   // double-buffered h
    float*  c_buf  = (float*) alloc((size_t)BATCH * HID * 4);
    __bf16* h1seq  = (__bf16*)alloc((size_t)SEQ * BATCH * HID * 2); // 67 MB
    float*  xgf    = (float*) alloc((size_t)SEQ * 128 * 4 * 256 * 4); // 536 MB (reused)

    // ---- one-time prep: pack weights (bf16 fragment layout), fold biases ----
    pack_w_kernel<<<512, 256, 0, stream>>>(W_ih0, wp_ih0, DIN_,  4 * HID);
    pack_w_kernel<<<512, 256, 0, stream>>>(W_hh0, wp_hh0, HID,   4 * HID);
    pack_w_kernel<<<512, 256, 0, stream>>>(W_ih1, wp_ih1, HID,   4 * DOUT_);
    pack_w_kernel<<<512, 256, 0, stream>>>(W_hh1, wp_hh1, DOUT_, 4 * DOUT_);
    combine_bias_kernel<<<(4 * HID   + 255) / 256, 256, 0, stream>>>(b_ih0, b_hh0, bias0, 4 * HID);
    combine_bias_kernel<<<(4 * DOUT_ + 255) / 256, 256, 0, stream>>>(b_ih1, b_hh1, bias1, 4 * DOUT_);

    const int R = SEQ * BATCH;  // 65536 GEMM rows

    // ---- layer 0 ----
    {
        const int Nsz = 4 * HID, Ksz = DIN_;
        const int npairs = (BATCH / 16) * (HID / 16);  // 128 waves
        const int blocks = (R / 16) * (Nsz / 16) / 8;  // one wave per 16x16 tile
        input_gemm_kernel<<<blocks, 256, 0, stream>>>(x, nullptr, wp_ih0, bias0, xgf,
                                                      Ksz, Nsz, npairs, 1);
        init_state_kernel<<<64, 256, 0, stream>>>(c_buf, h_buf, BATCH * HID,
                                                  2 * BATCH * HID, sync);
        lstm_rec_kernel<<<npairs / 8, 256, 0, stream>>>(xgf, wp_hh0, h_buf, c_buf,
                                                        h1seq, nullptr, HID, npairs,
                                                        sync, npairs / 8, 0);
    }

    // ---- layer 1 ----
    {
        const int Nsz = 4 * DOUT_, Ksz = HID;
        const int npairs = (BATCH / 16) * (DOUT_ / 16); // 64 waves
        const int blocks = (R / 16) * (Nsz / 16) / 8;
        input_gemm_kernel<<<blocks, 256, 0, stream>>>(nullptr, h1seq, wp_ih1, bias1, xgf,
                                                      Ksz, Nsz, npairs, 0);
        init_state_kernel<<<64, 256, 0, stream>>>(c_buf, h_buf, BATCH * DOUT_,
                                                  2 * BATCH * DOUT_, sync);
        lstm_rec_kernel<<<npairs / 8, 256, 0, stream>>>(xgf, wp_hh1, h_buf, c_buf,
                                                        nullptr, out, DOUT_, npairs,
                                                        sync, npairs / 8, 1);
    }
}
